// MultiQueryAttention_5884105195694
// MI455X (gfx1250) — compile-verified, matrix operand order checked by reading, not run on hardware
//
#include <hip/hip_runtime.h>
#include <hip/hip_bf16.h>

// ---------------------------------------------------------------------------
// MultiQueryAttention forward for MI455X (gfx1250, wave32, WMMA bf16)
// B=2, S=2048, D=2048, H=16, HD=128  (MQA: 1 KV head)
// ---------------------------------------------------------------------------

#define BATCH 2
#define SEQ   2048
#define DMODEL 2048
#define NHEAD 16
#define HDIM  128

typedef __bf16 bf16_t;
typedef __attribute__((ext_vector_type(16))) __bf16 v16bf;
typedef __attribute__((ext_vector_type(8)))  __bf16 v8bf;
typedef __attribute__((ext_vector_type(8)))  float  v8f;

union Frag16 {
    v16bf v;
    v8bf  h[2];
};

// ---------------------------------------------------------------------------
// fp32 -> bf16 elementwise convert
// ---------------------------------------------------------------------------
__global__ void mqa_f2bf_kernel(const float* __restrict__ in,
                                bf16_t* __restrict__ out, int n) {
    int i = blockIdx.x * blockDim.x + threadIdx.x;
    if (i < n) out[i] = (bf16_t)in[i];
}

// ---------------------------------------------------------------------------
// Generic WMMA GEMM:  C[M,N] (fp32) = A[M,K] (bf16, row-major)
//                                   x B[N,K]^T (bf16, row-major, K contiguous)
// One wave computes a 64x64 tile with a 4x4 grid of 16x16 WMMA accumulators.
// Block = (32,4): 4 waves take 4 adjacent N tiles.
// A-fragment layout per CDNA5 ISA: lane<16 -> M=lane, K runs {0..7,16..23};
// lane>=16 -> same M, K runs {8..15,24..31}.  B mirrors A with N=lane.
// ---------------------------------------------------------------------------
__global__ __launch_bounds__(128)
void mqa_gemm_bf16_nt_kernel(const bf16_t* __restrict__ A,
                             const bf16_t* __restrict__ Bm,
                             float* __restrict__ C,
                             int M, int N, int K) {
    const int lane = threadIdx.x;
    const int lid  = lane & 15;
    const int half = lane >> 4;
    const int ksel = half * 8;
    const int tileM = blockIdx.x * 64;
    const int tileN = (blockIdx.y * 4 + threadIdx.y) * 64;
    if (tileN >= N) return;             // wave-uniform exit (EXEC stays all-1s)

    v8f acc[4][4];
#pragma unroll
    for (int i = 0; i < 4; ++i)
#pragma unroll
        for (int j = 0; j < 4; ++j)
#pragma unroll
            for (int r = 0; r < 8; ++r) acc[i][j][r] = 0.0f;

    for (int k0 = 0; k0 < K; k0 += 32) {
        Frag16 a[4], b[4];
#pragma unroll
        for (int i = 0; i < 4; ++i) {
            const bf16_t* pa = A + (size_t)(tileM + 16 * i + lid) * K + k0 + ksel;
            a[i].h[0] = *(const v8bf*)pa;
            a[i].h[1] = *(const v8bf*)(pa + 16);
            const bf16_t* pb = Bm + (size_t)(tileN + 16 * i + lid) * K + k0 + ksel;
            b[i].h[0] = *(const v8bf*)pb;
            b[i].h[1] = *(const v8bf*)(pb + 16);
        }
#pragma unroll
        for (int i = 0; i < 4; ++i)
#pragma unroll
            for (int j = 0; j < 4; ++j)
                acc[i][j] = __builtin_amdgcn_wmma_f32_16x16x32_bf16(
                    false, a[i].v, false, b[j].v, (short)0, acc[i][j],
                    false, false);
    }

    // C/D layout: VGPR r, lanes 0-15 -> (M=r, N=lid); lanes 16-31 -> (M=r+8).
#pragma unroll
    for (int i = 0; i < 4; ++i) {
        const int rbase = tileM + 16 * i + half * 8;
#pragma unroll
        for (int j = 0; j < 4; ++j) {
            const int col = tileN + 16 * j + lid;
#pragma unroll
            for (int r = 0; r < 8; ++r)
                C[(size_t)(rbase + r) * N + col] = acc[i][j][r];
        }
    }
}

// ---------------------------------------------------------------------------
// RoPE on q: qf [B,S,D] fp32 (D = H*HD) -> qbf [B,H,S,HD] bf16
// pair index p in [0, HD/2); complex-pair rotation (view_as_complex layout)
// ---------------------------------------------------------------------------
__global__ void mqa_rope_q_kernel(const float* __restrict__ qf,
                                  const float* __restrict__ cosT,
                                  const float* __restrict__ sinT,
                                  bf16_t* __restrict__ qbf) {
    size_t pi = (size_t)blockIdx.x * blockDim.x + threadIdx.x;
    const size_t total = (size_t)BATCH * SEQ * NHEAD * (HDIM / 2);
    if (pi >= total) return;
    int p = pi % (HDIM / 2);
    int h = (pi / (HDIM / 2)) % NHEAD;
    int s = (pi / ((HDIM / 2) * NHEAD)) % SEQ;
    int b = pi / ((size_t)(HDIM / 2) * NHEAD * SEQ);
    float c  = cosT[s * (HDIM / 2) + p];
    float sn = sinT[s * (HDIM / 2) + p];
    size_t iin = ((size_t)(b * SEQ + s) * DMODEL) + h * HDIM + 2 * p;
    float x0 = qf[iin], x1 = qf[iin + 1];
    size_t iout = (((size_t)(b * NHEAD + h) * SEQ) + s) * HDIM + 2 * p;
    qbf[iout]     = (bf16_t)(x0 * c - x1 * sn);
    qbf[iout + 1] = (bf16_t)(x0 * sn + x1 * c);
}

// RoPE on k: kf [B,S,HD] fp32 -> kbf [B,S,HD] bf16
__global__ void mqa_rope_k_kernel(const float* __restrict__ kf,
                                  const float* __restrict__ cosT,
                                  const float* __restrict__ sinT,
                                  bf16_t* __restrict__ kbf) {
    size_t pi = (size_t)blockIdx.x * blockDim.x + threadIdx.x;
    const size_t total = (size_t)BATCH * SEQ * (HDIM / 2);
    if (pi >= total) return;
    int p = pi % (HDIM / 2);
    int s = (pi / (HDIM / 2)) % SEQ;
    int b = pi / ((size_t)(HDIM / 2) * SEQ);
    float c  = cosT[s * (HDIM / 2) + p];
    float sn = sinT[s * (HDIM / 2) + p];
    size_t idx = ((size_t)(b * SEQ + s) * HDIM) + 2 * p;
    float x0 = kf[idx], x1 = kf[idx + 1];
    kbf[idx]     = (bf16_t)(x0 * c - x1 * sn);
    kbf[idx + 1] = (bf16_t)(x0 * sn + x1 * c);
}

// v transpose: vf [B,S,HD] fp32 -> vT [B,HD,S] bf16 (for contiguous B-frags)
__global__ void mqa_vtrans_kernel(const float* __restrict__ vf,
                                  bf16_t* __restrict__ vT) {
    size_t i = (size_t)blockIdx.x * blockDim.x + threadIdx.x;
    const size_t total = (size_t)BATCH * SEQ * HDIM;
    if (i >= total) return;
    int d = i % HDIM;
    int s = (i / HDIM) % SEQ;
    int b = i / ((size_t)HDIM * SEQ);
    vT[((size_t)(b * HDIM + d)) * SEQ + s] =
        (bf16_t)vf[((size_t)(b * SEQ + s)) * HDIM + d];
}

// ---------------------------------------------------------------------------
// Causal flash attention (MQA). One wave per (b, h, 16-row q tile).
//   qbf [B,H,S,HD], kbf [B,S,HD], vT [B,HD,S]  (all bf16)
//   obf [B,S,H,HD] bf16
// Scores: QK^T via 4 chained 16x16x32 WMMAs per 16-key tile.
// Online softmax with half-wave shuffle row reductions.
// P staged through LDS (C-layout -> A-layout) for the PV WMMA.
// ---------------------------------------------------------------------------
__global__ __launch_bounds__(32)
void mqa_attn_kernel(const bf16_t* __restrict__ qbf,
                     const bf16_t* __restrict__ kbf,
                     const bf16_t* __restrict__ vT,
                     bf16_t* __restrict__ obf) {
    const int lane = threadIdx.x;
    const int lid  = lane & 15;
    const int half = lane >> 4;
    const int ksel = half * 8;
    const int qt   = blockIdx.x;            // q tile (16 rows)
    const int bh   = blockIdx.y;
    const int b = bh / NHEAD, h = bh % NHEAD;
    const int qrow0 = qt * 16;
    const float scale = 0.08838834764831843f;  // 1/sqrt(128)

    __shared__ bf16_t pbuf[16 * 32];        // 1 KB P-tile staging

    // Q fragments: 16 rows x 128 (4 k-chunks of 32)
    Frag16 qa[4];
    const bf16_t* qbase =
        qbf + (((size_t)(b * NHEAD + h) * SEQ) + qrow0 + lid) * HDIM;
#pragma unroll
    for (int kc = 0; kc < 4; ++kc) {
        qa[kc].h[0] = *(const v8bf*)(qbase + kc * 32 + ksel);
        qa[kc].h[1] = *(const v8bf*)(qbase + kc * 32 + ksel + 16);
    }

    v8f accO[8];
#pragma unroll
    for (int dt = 0; dt < 8; ++dt)
#pragma unroll
        for (int r = 0; r < 8; ++r) accO[dt][r] = 0.0f;

    float mrun[8], lrun[8];
#pragma unroll
    for (int r = 0; r < 8; ++r) { mrun[r] = -1e30f; lrun[r] = 0.0f; }

    const int nkb = (qrow0 + 15) / 32 + 1;  // causal key-block count (32 keys)
    for (int kb = 0; kb < nkb; ++kb) {
        const int key0 = kb * 32;

        // ---- scores: two 16x16 tiles over 32 keys ----
        v8f c[2];
#pragma unroll
        for (int jt = 0; jt < 2; ++jt) {
#pragma unroll
            for (int r = 0; r < 8; ++r) c[jt][r] = 0.0f;
            const bf16_t* kbase =
                kbf + ((size_t)(b * SEQ) + key0 + jt * 16 + lid) * HDIM;
#pragma unroll
            for (int kc = 0; kc < 4; ++kc) {
                Frag16 kf_;
                kf_.h[0] = *(const v8bf*)(kbase + kc * 32 + ksel);
                kf_.h[1] = *(const v8bf*)(kbase + kc * 32 + ksel + 16);
                c[jt] = __builtin_amdgcn_wmma_f32_16x16x32_bf16(
                    false, qa[kc].v, false, kf_.v, (short)0, c[jt],
                    false, false);
            }
        }

        // ---- online softmax per row (rows live in 16-lane groups) ----
#pragma unroll
        for (int r = 0; r < 8; ++r) {
            const int row = qrow0 + half * 8 + r;
            float s0 = c[0][r] * scale;
            float s1 = c[1][r] * scale;
            if (key0 + lid > row)      s0 = -1e30f;
            if (key0 + 16 + lid > row) s1 = -1e30f;
            float mx = fmaxf(s0, s1);
#pragma unroll
            for (int off = 8; off > 0; off >>= 1)
                mx = fmaxf(mx, __shfl_xor(mx, off, 32));
            const float newm = fmaxf(mrun[r], mx);
            const float corr = __expf(mrun[r] - newm);
            const float p0 = __expf(s0 - newm);
            const float p1 = __expf(s1 - newm);
            float rs = p0 + p1;
#pragma unroll
            for (int off = 8; off > 0; off >>= 1)
                rs += __shfl_xor(rs, off, 32);
            lrun[r] = lrun[r] * corr + rs;
            mrun[r] = newm;
#pragma unroll
            for (int dt = 0; dt < 8; ++dt) accO[dt][r] *= corr;
            const int row_l = half * 8 + r;
            pbuf[row_l * 32 + lid]      = (bf16_t)p0;
            pbuf[row_l * 32 + 16 + lid] = (bf16_t)p1;
        }
        __syncthreads();   // make P-tile (cross-lane) visible

        // reload P in A-matrix layout (16x32, contraction over keys)
        Frag16 pf;
        pf.h[0] = *(const v8bf*)(&pbuf[lid * 32 + ksel]);
        pf.h[1] = *(const v8bf*)(&pbuf[lid * 32 + ksel + 16]);
        __syncthreads();   // protect pbuf before next iteration's stores

        // ---- O += P x V : 8 hd tiles of 16 ----
#pragma unroll
        for (int dt = 0; dt < 8; ++dt) {
            const bf16_t* vb =
                vT + ((size_t)(b * HDIM) + dt * 16 + lid) * SEQ + key0;
            Frag16 vf_;
            vf_.h[0] = *(const v8bf*)(vb + ksel);
            vf_.h[1] = *(const v8bf*)(vb + ksel + 16);
            accO[dt] = __builtin_amdgcn_wmma_f32_16x16x32_bf16(
                false, pf.v, false, vf_.v, (short)0, accO[dt], false, false);
        }
    }

    // ---- normalize + store O as [B,S,H,HD] bf16 ----
#pragma unroll
    for (int r = 0; r < 8; ++r) {
        const float inv = 1.0f / lrun[r];
        const int row = qrow0 + half * 8 + r;
        bf16_t* ob = obf + (((size_t)(b * SEQ + row) * NHEAD) + h) * HDIM;
#pragma unroll
        for (int dt = 0; dt < 8; ++dt)
            ob[dt * 16 + lid] = (bf16_t)(accO[dt][r] * inv);
    }
}

// ---------------------------------------------------------------------------
// Launcher
// ---------------------------------------------------------------------------
extern "C" void kernel_launch(void* const* d_in, const int* in_sizes, int n_in,
                              void* d_out, int out_size, void* d_ws, size_t ws_size,
                              hipStream_t stream) {
    const float* x    = (const float*)d_in[0];  // [B,S,D]
    const float* Wq   = (const float*)d_in[1];  // [D,D]
    const float* Wk   = (const float*)d_in[2];  // [HD,D]
    const float* Wv   = (const float*)d_in[3];  // [HD,D]
    const float* Wo   = (const float*)d_in[4];  // [D,D]
    const float* cosT = (const float*)d_in[5];  // [S,HD/2]
    const float* sinT = (const float*)d_in[6];  // [S,HD/2]
    float* out = (float*)d_out;                 // [B,S,D]

    char* ws = (char*)d_ws;
    const size_t MROWS = (size_t)BATCH * SEQ;   // 4096
    // workspace layout (all offsets 256B aligned)
    size_t o_xbf = 0;                                          // 16 MB
    size_t o_wq  = o_xbf + MROWS * DMODEL * 2;                 //  8 MB
    size_t o_wk  = o_wq  + (size_t)DMODEL * DMODEL * 2;        // .5 MB
    size_t o_wv  = o_wk  + (size_t)HDIM * DMODEL * 2;          // .5 MB
    size_t o_wo  = o_wv  + (size_t)HDIM * DMODEL * 2;          //  8 MB
    size_t o_qf  = o_wo  + (size_t)DMODEL * DMODEL * 2;        // 32 MB fp32
    size_t o_kf  = o_qf  + MROWS * DMODEL * 4;                 //  2 MB fp32
    size_t o_vf  = o_kf  + MROWS * HDIM * 4;                   //  2 MB fp32
    size_t o_qbf = o_vf  + MROWS * HDIM * 4;                   // 16 MB
    size_t o_kbf = o_qbf + MROWS * DMODEL * 2;                 //  1 MB
    size_t o_vT  = o_kbf + MROWS * HDIM * 2;                   //  1 MB
    size_t o_obf = o_qf;  // alias: qf is dead after rope_q

    bf16_t* xbf = (bf16_t*)(ws + o_xbf);
    bf16_t* wqb = (bf16_t*)(ws + o_wq);
    bf16_t* wkb = (bf16_t*)(ws + o_wk);
    bf16_t* wvb = (bf16_t*)(ws + o_wv);
    bf16_t* wob = (bf16_t*)(ws + o_wo);
    float*  qf  = (float*) (ws + o_qf);
    float*  kf  = (float*) (ws + o_kf);
    float*  vf  = (float*) (ws + o_vf);
    bf16_t* qbf = (bf16_t*)(ws + o_qbf);
    bf16_t* kbf = (bf16_t*)(ws + o_kbf);
    bf16_t* vT  = (bf16_t*)(ws + o_vT);
    bf16_t* obf = (bf16_t*)(ws + o_obf);

    // 1) converts to bf16
    {
        int n;
        n = (int)(MROWS * DMODEL);
        mqa_f2bf_kernel<<<(n + 255) / 256, 256, 0, stream>>>(x, xbf, n);
        n = DMODEL * DMODEL;
        mqa_f2bf_kernel<<<(n + 255) / 256, 256, 0, stream>>>(Wq, wqb, n);
        n = HDIM * DMODEL;
        mqa_f2bf_kernel<<<(n + 255) / 256, 256, 0, stream>>>(Wk, wkb, n);
        mqa_f2bf_kernel<<<(n + 255) / 256, 256, 0, stream>>>(Wv, wvb, n);
        n = DMODEL * DMODEL;
        mqa_f2bf_kernel<<<(n + 255) / 256, 256, 0, stream>>>(Wo, wob, n);
    }

    // 2) projections: y = x @ W^T  (A=[M,K], B=[N,K] both K-contiguous)
    {
        dim3 blk(32, 4);
        dim3 gq((int)MROWS / 64, (DMODEL / 64 + 3) / 4);   // (64, 8)
        mqa_gemm_bf16_nt_kernel<<<gq, blk, 0, stream>>>(xbf, wqb, qf,
                                                        (int)MROWS, DMODEL, DMODEL);
        dim3 gk((int)MROWS / 64, (HDIM / 64 + 3) / 4);     // (64, 1)
        mqa_gemm_bf16_nt_kernel<<<gk, blk, 0, stream>>>(xbf, wkb, kf,
                                                        (int)MROWS, HDIM, DMODEL);
        mqa_gemm_bf16_nt_kernel<<<gk, blk, 0, stream>>>(xbf, wvb, vf,
                                                        (int)MROWS, HDIM, DMODEL);
    }

    // 3) RoPE + layout repack to bf16
    {
        int n;
        n = BATCH * SEQ * NHEAD * (HDIM / 2);
        mqa_rope_q_kernel<<<(n + 255) / 256, 256, 0, stream>>>(qf, cosT, sinT, qbf);
        n = BATCH * SEQ * (HDIM / 2);
        mqa_rope_k_kernel<<<(n + 255) / 256, 256, 0, stream>>>(kf, cosT, sinT, kbf);
        n = BATCH * SEQ * HDIM;
        mqa_vtrans_kernel<<<(n + 255) / 256, 256, 0, stream>>>(vf, vT);
    }

    // 4) causal flash attention (one wave per 16-row q tile per (b,h))
    {
        dim3 g(SEQ / 16, BATCH * NHEAD);  // (128, 32)
        mqa_attn_kernel<<<g, 32, 0, stream>>>(qbf, kbf, vT, obf);
    }

    // 5) output projection -> d_out (fp32)
    {
        dim3 blk(32, 4);
        dim3 go((int)MROWS / 64, (DMODEL / 64 + 3) / 4);
        mqa_gemm_bf16_nt_kernel<<<go, blk, 0, stream>>>(obf, wob, out,
                                                        (int)MROWS, DMODEL, DMODEL);
    }
}
